// CRF_70239895159020
// MI455X (gfx1250) — matrix operand order repaired and checked by reading, not compiled, not run on hardware
//
#include <hip/hip_runtime.h>
#include <hip/hip_bf16.h>

typedef __attribute__((ext_vector_type(16))) _Float16 v16h;
typedef __attribute__((ext_vector_type(8)))  _Float16 v8h;
typedef __attribute__((ext_vector_type(8)))  float    v8f;

#define C_DIM    70
#define NT       5     // label tiles of 16 -> 80 (padded)
#define NK       3     // K steps of 32 -> 96 (padded)
#define LSTRIDE  104   // halves per L row (bank-spread pad, 16B aligned)
#define PSTRIDE  104   // halves per P row
#define ROWS_PER_BLK 128
#define N_ITERS  10

__device__ __forceinline__ float fast_tanh(float x) {
#if __has_builtin(__builtin_amdgcn_tanhf)
    return __builtin_amdgcn_tanhf(x);
#else
    float y;
    // v_tanh_f32 is a TRANS op; v_nop covers the multicycle co-exec hazard
    asm volatile("v_tanh_f32 %0, %1\n\tv_nop" : "=v"(y) : "v"(x));
    return y;
#endif
}

__device__ __forceinline__ v16h cat16(v8h lo, v8h hi) {
    return __builtin_shufflevector(lo, hi, 0,1,2,3,4,5,6,7,8,9,10,11,12,13,14,15);
}

// Computes 10 mean-field CRF iterations, fully fused.
//   tmp^T = L0^T @ P^T  (+ logits), L0 = 0.5*(S+S^T) with zero diagonal
//   P <- tanh(tmp);  final Q = log_sigmoid(2*tmp)
__global__ __launch_bounds__(256)
void crf_meanfield_kernel(const float* __restrict__ logits,
                          const float* __restrict__ sims,
                          float* __restrict__ out, int B) {
    __shared__ __align__(16) _Float16 Lsh[80 * LSTRIDE];      // L (symmetric), diag+pads zeroed
    __shared__ __align__(16) _Float16 Psh[8 * 16 * PSTRIDE];  // per-wave P: 16 rows x 96 K (f16)

    const int tid = threadIdx.x;

    // ---- stage L0 into LDS, zero-padded to 80x96, zero diagonal ----
    for (int idx = tid; idx < 80 * LSTRIDE; idx += 256) {
        int n = idx / LSTRIDE, k = idx - n * LSTRIDE;
        float v = 0.0f;
        if (n < C_DIM && k < C_DIM && n != k)
            v = 0.5f * (sims[n * C_DIM + k] + sims[k * C_DIM + n]);
        Lsh[idx] = (_Float16)v;
    }
    // zero P tiles (K columns 80..95 must stay zero; 70..79 are killed by zero L rows)
    for (int idx = tid; idx < 8 * 16 * PSTRIDE; idx += 256)
        Psh[idx] = (_Float16)0.0f;
    __syncthreads();

    const int warp = tid >> 5, lane = tid & 31;
    const int hi = lane >> 4, ln = lane & 15;
    const long row = (long)blockIdx.x * ROWS_PER_BLK + warp * 16 + ln;  // this lane's batch row
    _Float16* pw = &Psh[warp * 16 * PSTRIDE];

    // ---- hoist all A-matrix (L^T) WMMA operands into registers (loop-invariant) ----
    // A tile (t,ks): rows = labels 16t..16t+15, K = 32ks..32ks+31
    // A layout: lane m=ln, VGPR0-3 <- K=kb..kb+7, VGPR4-7 <- K=kb+16..kb+23, kb=32ks+8*hi
    v16h La[NK][NT];
#pragma unroll
    for (int ks = 0; ks < NK; ++ks)
#pragma unroll
        for (int t = 0; t < NT; ++t) {
            const _Float16* ap = &Lsh[(t * 16 + ln) * LSTRIDE + ks * 32 + hi * 8];
            La[ks][t] = cat16(*(const v8h*)ap, *(const v8h*)(ap + 16));
        }

    // ---- load logits in D layout: lane holds labels c = 16t + 8*hi + v of its row ----
    float lg[NT][8];
#pragma unroll
    for (int t = 0; t < NT; ++t)
#pragma unroll
        for (int p = 0; p < 4; ++p) {
            const int c = 16 * t + 8 * hi + 2 * p;
            if (c < C_DIM && row < B) {
                const float2 v = *(const float2*)(logits + row * C_DIM + c);
                lg[t][2 * p] = v.x; lg[t][2 * p + 1] = v.y;
            } else {
                lg[t][2 * p] = -1e30f; lg[t][2 * p + 1] = -1e30f;
            }
        }

    // ---- P0 = 2*softmax(logits) - 1 ; row reduction = in-lane + one xor-16 shuffle ----
    {
        float m = -1e30f;
#pragma unroll
        for (int t = 0; t < NT; ++t)
#pragma unroll
            for (int v = 0; v < 8; ++v) m = fmaxf(m, lg[t][v]);
        m = fmaxf(m, __shfl_xor(m, 16, 32));
        float e[NT][8], s = 0.0f;
#pragma unroll
        for (int t = 0; t < NT; ++t)
#pragma unroll
            for (int v = 0; v < 8; ++v) { e[t][v] = __expf(lg[t][v] - m); s += e[t][v]; }
        s += __shfl_xor(s, 16, 32);
        const float inv = 1.0f / s;
#pragma unroll
        for (int t = 0; t < NT; ++t) {
            v8h ph;
#pragma unroll
            for (int v = 0; v < 8; ++v)
                ph[v] = (_Float16)(2.0f * e[t][v] * inv - 1.0f);
            *(v8h*)&pw[ln * PSTRIDE + t * 16 + hi * 8] = ph;   // contiguous 16B store
        }
    }
    // No barriers in the loop: Lsh is read-only now; each wave touches only its own P tile
    // and LDS ops are in-order within a wave.

#pragma unroll 1
    for (int it = 0; it < N_ITERS; ++it) {
        v8f acc[NT];
#pragma unroll
        for (int t = 0; t < NT; ++t)
#pragma unroll
            for (int v = 0; v < 8; ++v) acc[t][v] = lg[t][v];

#pragma unroll
        for (int ks = 0; ks < NK; ++ks) {
            // B = P^T 32x16: lane col = own row ln; K=kb+8*hi..+7 and kb+16+8*hi..+7
            const _Float16* bp = &pw[ln * PSTRIDE + ks * 32 + hi * 8];
            const v16h Bm = cat16(*(const v8h*)bp, *(const v8h*)(bp + 16));
#pragma unroll
            for (int t = 0; t < NT; ++t)
                acc[t] = __builtin_amdgcn_wmma_f32_16x16x32_f16(
                    false, La[ks][t], false, Bm, (short)0, acc[t], false, false);
        }

        if (it + 1 < N_ITERS) {
            // P <- tanh(tmp): 8 contiguous labels per lane per tile -> one b128 store each
#pragma unroll
            for (int t = 0; t < NT; ++t) {
                v8h ph;
#pragma unroll
                for (int v = 0; v < 8; ++v)
                    ph[v] = (_Float16)fast_tanh(acc[t][v]);
                *(v8h*)&pw[ln * PSTRIDE + t * 16 + hi * 8] = ph;
            }
        } else {
            // Q = log_sigmoid(2*tmp) = min(2t,0) - log1p(exp(-|2t|)); paired f32 stores
#pragma unroll
            for (int t = 0; t < NT; ++t)
#pragma unroll
                for (int p = 0; p < 4; ++p) {
                    const int c = 16 * t + 8 * hi + 2 * p;
                    if (c < C_DIM && row < B) {
                        const float ta = 2.0f * acc[t][2 * p];
                        const float tb = 2.0f * acc[t][2 * p + 1];
                        float2 q;
                        q.x = fminf(ta, 0.0f) - __logf(1.0f + __expf(-fabsf(ta)));
                        q.y = fminf(tb, 0.0f) - __logf(1.0f + __expf(-fabsf(tb)));
                        *(float2*)(out + row * C_DIM + c) = q;
                    }
                }
        }
    }
}

extern "C" void kernel_launch(void* const* d_in, const int* in_sizes, int n_in,
                              void* d_out, int out_size, void* d_ws, size_t ws_size,
                              hipStream_t stream) {
    (void)n_in; (void)out_size; (void)d_ws; (void)ws_size;
    const float* logits = (const float*)d_in[0];
    const float* sims   = (const float*)d_in[1];
    float* out          = (float*)d_out;
    const int B = in_sizes[0] / C_DIM;                        // 524288
    const int grid = (B + ROWS_PER_BLK - 1) / ROWS_PER_BLK;   // 4096
    crf_meanfield_kernel<<<grid, 256, 0, stream>>>(logits, sims, out, B);
}
